// LNO_1675037245694
// MI455X (gfx1250) — compile-verified, matrix operand order checked by reading, not run on hardware
//
#include <hip/hip_runtime.h>

#define K_MODES 20
#define HDIM    128
#define NF      16
#define DIMS    3
#define LIFT_IN 22
#define OUTD    4
#define NLAYERS 4

typedef __attribute__((ext_vector_type(8)))  float  v8f;
typedef __attribute__((ext_vector_type(8)))  __bf16 v8bf;
typedef __attribute__((ext_vector_type(16))) __bf16 v16bf;

static __device__ __forceinline__ unsigned short f2bf(float f) {
    unsigned int u = __builtin_bit_cast(unsigned int, f);
    u += 0x7FFFu + ((u >> 16) & 1u);   // round-to-nearest-even
    return (unsigned short)(u >> 16);
}
static __device__ __forceinline__ float bf2f(unsigned short h) {
    unsigned int u = ((unsigned int)h) << 16;
    return __builtin_bit_cast(float, u);
}
static __device__ __forceinline__ float elu1(float x) {
    return x > 0.f ? x : (__expf(x) - 1.f);
}

// ---------------------------------------------------------------- G = phi^T phi
__global__ void __launch_bounds__(256) k_gram(const float* __restrict__ phi,
                                              float* __restrict__ G, int N) {
    __shared__ float Gs[K_MODES * K_MODES];
    for (int i = threadIdx.x; i < K_MODES * K_MODES; i += 256) Gs[i] = 0.f;
    __syncthreads();
    for (int n = blockIdx.x * 256 + threadIdx.x; n < N; n += gridDim.x * 256) {
        float p[K_MODES];
#pragma unroll
        for (int k = 0; k < K_MODES; k++) p[k] = phi[n * K_MODES + k];
#pragma unroll
        for (int i = 0; i < K_MODES; i++)
#pragma unroll
            for (int j = i; j < K_MODES; j++)
                atomicAdd(&Gs[i * K_MODES + j], p[i] * p[j]);
    }
    __syncthreads();
    for (int i = threadIdx.x; i < K_MODES * K_MODES; i += 256) {
        int r = i / K_MODES, c = i % K_MODES;
        if (c >= r) {
            atomicAdd(&G[i], Gs[i]);
            if (c > r) atomicAdd(&G[c * K_MODES + r], Gs[i]);
        }
    }
}

// ------------------------------------------------- Ginv = inv(G) (Gauss-Jordan)
__global__ void __launch_bounds__(64) k_inv(const float* __restrict__ G,
                                            float* __restrict__ Gi) {
    __shared__ float A[K_MODES][2 * K_MODES];
    __shared__ float f[K_MODES];
    const int t = threadIdx.x;
    for (int i = t; i < K_MODES * 2 * K_MODES; i += 64) {
        int r = i / (2 * K_MODES), c = i % (2 * K_MODES);
        A[r][c] = (c < K_MODES) ? G[r * K_MODES + c] : ((c - K_MODES == r) ? 1.f : 0.f);
    }
    __syncthreads();
    for (int col = 0; col < K_MODES; col++) {
        float piv = A[col][col];
        __syncthreads();
        float inv = 1.f / piv;
        for (int c = t; c < 2 * K_MODES; c += 64) A[col][c] *= inv;
        __syncthreads();
        if (t < K_MODES) f[t] = (t == col) ? 0.f : A[t][col];
        __syncthreads();
        for (int i = t; i < K_MODES * 2 * K_MODES; i += 64) {
            int r = i / (2 * K_MODES), c = i % (2 * K_MODES);
            A[r][c] -= f[r] * A[col][c];
        }
        __syncthreads();
    }
    for (int i = t; i < K_MODES * K_MODES; i += 64)
        Gi[i] = A[i / K_MODES][K_MODES + i % K_MODES];
}

// --------------------------------------------------------- f32 -> bf16 convert
__global__ void k_conv(const float* __restrict__ src,
                       unsigned short* __restrict__ dst, int n) {
    int i = blockIdx.x * blockDim.x + threadIdx.x;
    if (i < n) dst[i] = f2bf(src[i]);
}

// ----------------------------------------------------------------- lift kernel
__global__ void __launch_bounds__(256) k_lift(
    const float* __restrict__ x, const float* __restrict__ pos,
    const float* __restrict__ y, const float* __restrict__ lw,
    const float* __restrict__ lb, unsigned short* __restrict__ h, int N) {
    __shared__ float Wsh[HDIM * LIFT_IN];
    __shared__ float insh[16 * LIFT_IN];
    const int t = threadIdx.x;
    const int base = blockIdx.x * 16;
    for (int i = t; i < HDIM * LIFT_IN; i += 256) Wsh[i] = lw[i];
    for (int i = t; i < 16 * LIFT_IN; i += 256) {
        int r = i / LIFT_IN, fidx = i % LIFT_IN;
        int n = base + r;
        float v = 0.f;
        if (n < N) {
            if (fidx < NF)             v = x[(size_t)n * NF + fidx];
            else if (fidx < NF + DIMS) v = pos[(size_t)n * DIMS + (fidx - NF)];
            else                       v = y[fidx - NF - DIMS];
        }
        insh[i] = v;
    }
    __syncthreads();
    const int j = t & (HDIM - 1);
    const int half = t >> 7;
    const float bias = lb[j];
#pragma unroll
    for (int s = 0; s < 8; s++) {
        int r = half * 8 + s;
        float acc = bias;
#pragma unroll
        for (int fidx = 0; fidx < LIFT_IN; fidx++)
            acc += insh[r * LIFT_IN + fidx] * Wsh[j * LIFT_IN + fidx];
        h[(size_t)(base + r) * HDIM + j] = f2bf(acc);
    }
}

// ----------------------------------------------------- PtH[k][j] = sum phi * h
__global__ void __launch_bounds__(128) k_red(const unsigned short* __restrict__ h,
                                             const float* __restrict__ phi,
                                             float* __restrict__ PtH, int N) {
    __shared__ float ps[32 * K_MODES];
    const int j = threadIdx.x;
    float acc[K_MODES];
#pragma unroll
    for (int k = 0; k < K_MODES; k++) acc[k] = 0.f;
    for (int nb = blockIdx.x * 32; nb < N; nb += gridDim.x * 32) {
        for (int i = j; i < 32 * K_MODES; i += 128) {
            int n = nb + i / K_MODES;
            ps[i] = (n < N) ? phi[(size_t)n * K_MODES + i % K_MODES] : 0.f;
        }
        __syncthreads();
#pragma unroll 4
        for (int s = 0; s < 32; s++) {
            int n = nb + s;
            float hv = (n < N) ? bf2f(h[(size_t)n * HDIM + j]) : 0.f;
#pragma unroll
            for (int k = 0; k < K_MODES; k++) acc[k] += ps[s * K_MODES + k] * hv;
        }
        __syncthreads();
    }
#pragma unroll
    for (int k = 0; k < K_MODES; k++) atomicAdd(&PtH[k * HDIM + j], acc[k]);
}

// -------------------------- v = Ginv @ PtH ; RV[k][i] = kern[k][i][:] . v[k][:]
__global__ void __launch_bounds__(128) k_spec(const float* __restrict__ Ginv,
                                              const float* __restrict__ PtH,
                                              const float* __restrict__ kern,
                                              float* __restrict__ RV) {
    const int k = blockIdx.x;
    const int i = threadIdx.x;
    __shared__ float v[HDIM];
    float acc = 0.f;
#pragma unroll
    for (int kk = 0; kk < K_MODES; kk++)
        acc += Ginv[k * K_MODES + kk] * PtH[kk * HDIM + i];
    v[i] = acc;
    __syncthreads();
    const float* kr = kern + ((size_t)k * HDIM + i) * HDIM;
    float r = 0.f;
#pragma unroll 8
    for (int m = 0; m < HDIM; m++) r += kr[m] * v[m];
    RV[k * HDIM + i] = r;
}

// ---- main WMMA layer: hout = elu([hin | phi] @ [W | rv]^T), K=160 all-WMMA ----
__global__ void __launch_bounds__(256) k_layer(
    const unsigned short* __restrict__ hin,      // [Npad,H] bf16
    unsigned short* __restrict__ hout,           // [Npad,H] bf16
    const unsigned short* __restrict__ linw_bf,  // [H,H] bf16 (this layer)
    const float* __restrict__ phi,               // [N,K]
    const float* __restrict__ RV,                // [K,H]
    int N, int Npad) {
    __shared__ __align__(16) unsigned short Bsh[HDIM * HDIM];  // 32 KB  W[j][m]
    __shared__ __align__(16) unsigned short RVT[HDIM * 32];    // 8 KB   rv^T[j][k] pad k->32
    __shared__ __align__(16) unsigned short PhiB[128 * 32];    // 8 KB   phi[r][k] pad k->32

    const int t = threadIdx.x;
    const int nb_block = blockIdx.x * 128;

    {   // stage layer weights (bf16) into LDS
        uint4* dst = (uint4*)Bsh;
        const uint4* src = (const uint4*)linw_bf;
        for (int i = t; i < HDIM * HDIM / 8; i += 256) dst[i] = src[i];
    }
    for (int i = t; i < HDIM * 32; i += 256) {
        int j = i >> 5, k = i & 31;
        RVT[i] = (k < K_MODES) ? f2bf(RV[k * HDIM + j]) : (unsigned short)0;
    }
    for (int i = t; i < 128 * 32; i += 256) {
        int r = i >> 5, k = i & 31;
        int n = nb_block + r;
        PhiB[i] = (n < N && k < K_MODES) ? f2bf(phi[(size_t)n * K_MODES + k])
                                         : (unsigned short)0;
    }
    __syncthreads();

    const int wave = t >> 5;
    const int lane = t & 31;
    const int rr   = lane & 15;
    const int hi   = lane >> 4;
    const int nb   = nb_block + wave * 16;

    // prefetch next block's A rows (global_prefetch_b8)
    if (nb_block + 128 < Npad)
        __builtin_prefetch(hin + (size_t)(nb + 128 + rr) * HDIM, 0, 1);

    // A fragments: 4 k-steps from h (global bf16) + 1 k-step from phi (LDS)
    v16bf a[5];
    const unsigned short* arow = hin + (size_t)(nb + rr) * HDIM;
#pragma unroll
    for (int ks = 0; ks < 4; ks++) {
        const v8bf lo = *(const v8bf*)(arow + ks * 32 + 8 * hi);
        const v8bf hb = *(const v8bf*)(arow + ks * 32 + 16 + 8 * hi);
#pragma unroll
        for (int e = 0; e < 8; e++) { a[ks][e] = lo[e]; a[ks][8 + e] = hb[e]; }
    }
    {
        const unsigned short* prow = PhiB + (wave * 16 + rr) * 32;
        const v8bf lo = *(const v8bf*)(prow + 8 * hi);
        const v8bf hb = *(const v8bf*)(prow + 16 + 8 * hi);
#pragma unroll
        for (int e = 0; e < 8; e++) { a[4][e] = lo[e]; a[4][8 + e] = hb[e]; }
    }

#pragma unroll
    for (int jt = 0; jt < 8; jt++) {
        const int jb = jt * 16;
        v8f c = {};
        // k-step 4: phi @ rv^T tile
        {
            const unsigned short* rvrow = RVT + (jb + rr) * 32;
            v16bf b;
            const v8bf lo = *(const v8bf*)(rvrow + 8 * hi);
            const v8bf hb = *(const v8bf*)(rvrow + 16 + 8 * hi);
#pragma unroll
            for (int e = 0; e < 8; e++) { b[e] = lo[e]; b[8 + e] = hb[e]; }
            c = __builtin_amdgcn_wmma_f32_16x16x32_bf16(
                    false, a[4], false, b, (short)0, c, false, false);
        }
        // k-steps 0..3: h @ W^T over K=128
        const unsigned short* brow = Bsh + (jb + rr) * HDIM;
#pragma unroll
        for (int ks = 0; ks < 4; ks++) {
            v16bf b;
            const v8bf lo = *(const v8bf*)(brow + ks * 32 + 8 * hi);
            const v8bf hb = *(const v8bf*)(brow + ks * 32 + 16 + 8 * hi);
#pragma unroll
            for (int e = 0; e < 8; e++) { b[e] = lo[e]; b[8 + e] = hb[e]; }
            c = __builtin_amdgcn_wmma_f32_16x16x32_bf16(
                    false, a[ks], false, b, (short)0, c, false, false);
        }
        // ELU + bf16 store
#pragma unroll
        for (int g = 0; g < 8; g++) {
            const int m = g + 8 * hi;
            hout[(size_t)(nb + m) * HDIM + jb + rr] = f2bf(elu1(c[g]));
        }
    }
}

// -------------------------------------------------------------- final project
__global__ void __launch_bounds__(256) k_lower(const unsigned short* __restrict__ h,
                                               const float* __restrict__ lw,
                                               const float* __restrict__ lb,
                                               float* __restrict__ out, int N) {
    __shared__ float Wsh[OUTD * HDIM];
    __shared__ float bsh[OUTD];
    const int t = threadIdx.x;
    for (int i = t; i < OUTD * HDIM; i += 256) Wsh[i] = lw[i];
    if (t < OUTD) bsh[t] = lb[t];
    __syncthreads();
    int n = blockIdx.x * 256 + t;
    if (n >= N) return;
    const unsigned short* hr = h + (size_t)n * HDIM;
    float a0 = 0.f, a1 = 0.f, a2 = 0.f, a3 = 0.f;
#pragma unroll 8
    for (int j = 0; j < HDIM; j++) {
        float hv = bf2f(hr[j]);
        a0 += hv * Wsh[0 * HDIM + j];
        a1 += hv * Wsh[1 * HDIM + j];
        a2 += hv * Wsh[2 * HDIM + j];
        a3 += hv * Wsh[3 * HDIM + j];
    }
    out[(size_t)n * OUTD + 0] = a0 + bsh[0];
    out[(size_t)n * OUTD + 1] = a1 + bsh[1];
    out[(size_t)n * OUTD + 2] = a2 + bsh[2];
    out[(size_t)n * OUTD + 3] = a3 + bsh[3];
}

extern "C" void kernel_launch(void* const* d_in, const int* in_sizes, int n_in,
                              void* d_out, int out_size, void* d_ws, size_t ws_size,
                              hipStream_t stream) {
    const float* x       = (const float*)d_in[0];
    const float* pos     = (const float*)d_in[1];
    const float* y       = (const float*)d_in[2];
    const float* phi     = (const float*)d_in[3];
    const float* lift_w  = (const float*)d_in[4];
    const float* lift_b  = (const float*)d_in[5];
    const float* lin_w   = (const float*)d_in[6];
    const float* kern    = (const float*)d_in[7];
    const float* lower_w = (const float*)d_in[8];
    const float* lower_b = (const float*)d_in[9];
    float* out = (float*)d_out;

    const int N    = in_sizes[0] / NF;           // 200000
    const int Npad = (N + 127) & ~127;           // 200064

    char* ws = (char*)d_ws;
    size_t off = 0;
    auto alloc = [&](size_t bytes) {
        void* p = ws + off;
        off = (off + bytes + 255) & ~(size_t)255;
        return p;
    };
    unsigned short* hA      = (unsigned short*)alloc((size_t)Npad * HDIM * 2);
    unsigned short* hB      = (unsigned short*)alloc((size_t)Npad * HDIM * 2);
    unsigned short* linw_bf = (unsigned short*)alloc((size_t)NLAYERS * HDIM * HDIM * 2);
    float* G    = (float*)alloc(K_MODES * K_MODES * 4);
    float* Ginv = (float*)alloc(K_MODES * K_MODES * 4);
    float* PtH  = (float*)alloc(K_MODES * HDIM * 4);
    float* RV   = (float*)alloc(K_MODES * HDIM * 4);
    (void)ws_size; (void)out_size; (void)n_in;

    hipMemsetAsync(G, 0, K_MODES * K_MODES * 4, stream);
    k_gram<<<256, 256, 0, stream>>>(phi, G, N);
    k_inv<<<1, 64, 0, stream>>>(G, Ginv);
    {
        int n = NLAYERS * HDIM * HDIM;
        k_conv<<<(n + 255) / 256, 256, 0, stream>>>(lin_w, linw_bf, n);
    }
    k_lift<<<Npad / 16, 256, 0, stream>>>(x, pos, y, lift_w, lift_b, hA, N);

    unsigned short* cur = hA;
    unsigned short* nxt = hB;
    for (int l = 0; l < NLAYERS; l++) {
        hipMemsetAsync(PtH, 0, K_MODES * HDIM * 4, stream);
        k_red<<<512, 128, 0, stream>>>(cur, phi, PtH, N);
        k_spec<<<K_MODES, HDIM, 0, stream>>>(
            Ginv, PtH, kern + (size_t)l * K_MODES * HDIM * HDIM, RV);
        k_layer<<<Npad / 128, 256, 0, stream>>>(
            cur, nxt, linw_bf + (size_t)l * HDIM * HDIM, phi, RV, N, Npad);
        unsigned short* tmp = cur; cur = nxt; nxt = tmp;
    }
    k_lower<<<(N + 255) / 256, 256, 0, stream>>>(cur, lower_w, lower_b, out, N);
}
